// E2AttentionArbOrder_sparse_forcluster_9577777070592
// MI455X (gfx1250) — matrix operand order repaired and checked by reading, not compiled, not run on hardware
//
#include <hip/hip_runtime.h>
#include <hip/hip_bf16.h>

// ---- problem constants ------------------------------------------------------
#define NN   2048
#define MMc  2048
#define KK   32
#define CCd  256
#define AWW  32
#define NEE  128
#define HHh  8
#define SSs  32
#define ECC  416
#define M0D  192
#define HIDD 16

typedef __attribute__((ext_vector_type(16))) _Float16 v16h;
typedef __attribute__((ext_vector_type(8)))  _Float16 v8h;
typedef __attribute__((ext_vector_type(8)))  float    v8f;

// ---- WMMA helpers (A/B in contiguous k = 16h + s order) ----------------------------
__device__ __forceinline__ v8f wmma16(v16h a, v16h b, v8f c) {
    return __builtin_amdgcn_wmma_f32_16x16x32_f16(false, a, false, b, (short)0, c,
                                                  false, false);
}

// A tile (16x32 f16, row-major, stride lda, lda % 8 == 0, k0 % 32 == 0, full K).
// lane&15 = M row; halves j -> K = k0+g+j; g=16*(lane>=16).
// One 32B vector load per lane.
__device__ __forceinline__ v16h load_a_h_full(const _Float16* __restrict__ src, int lda,
                                              int row0, int k0, int lane) {
    const _Float16* p = src + (size_t)(row0 + (lane & 15)) * lda + k0 + ((lane & 16) ? 16 : 0);
    return *(const v16h*)p;
}

// A tile with K padded to 32 (kmax < 32, k0 = 0). Scalar guarded.
__device__ __forceinline__ v16h load_a_h_pad(const _Float16* __restrict__ src, int lda,
                                             int row0, int kmax, int lane) {
    const int g = (lane & 16) ? 16 : 0;
    const _Float16* p = src + (size_t)(row0 + (lane & 15)) * lda;
    v16h a;
#pragma unroll
    for (int j = 0; j < 16; ++j) {
        int kk = g + j;
        a[j] = (kk < kmax) ? p[kk] : (_Float16)0.0f;
    }
    return a;
}

// B tile (32x16) with B[k][n] = W[n][k], W row-major f16 [ND x KD], KD % 16 == 0,
// k0 % 32 == 0, full K. lane&15 = N col, halves = 16 contiguous K -> one 32B load.
__device__ __forceinline__ v16h load_b_h_full(const _Float16* __restrict__ W, int kd,
                                              int nd0, int k0, int lane) {
    const _Float16* p = W + (size_t)(nd0 + (lane & 15)) * kd + k0 + ((lane & 16) ? 16 : 0);
    return *(const v16h*)p;
}

// B tile with K padded to 32 (kmax < 32, k0 = 0). Scalar guarded.
__device__ __forceinline__ v16h load_b_h_pad(const _Float16* __restrict__ W, int kd,
                                             int nd0, int kmax, int lane) {
    const int kb = (lane & 16) ? 16 : 0;
    const _Float16* p = W + (size_t)(nd0 + (lane & 15)) * kd + kb;
    v16h b;
#pragma unroll
    for (int j = 0; j < 16; ++j) {
        int kk = kb + j;
        b[j] = (kk < kmax) ? p[j] : (_Float16)0.0f;
    }
    return b;
}

__device__ __forceinline__ int l_of_m(int m) { return (m == 0) ? 0 : ((m < 4) ? 1 : 2); }

// ---- Kernel 0: f32 -> f16 conversion (weights + big A-side arrays) ----------
__global__ __launch_bounds__(256) void k_cvt(const float* __restrict__ src,
                                             _Float16* __restrict__ dst, int n) {
    int i = blockIdx.x * 256 + threadIdx.x;
    if (i < n) dst[i] = (_Float16)src[i];
}

// ---- Kernel 1: q = so3_linear(node), key = so3_linear(cluster) --------------
// grid: (256, 9)  block: 32.  Combined N+M row tiles; per-m weight select.
__global__ __launch_bounds__(32) void k_proj(const _Float16* __restrict__ node_h,
                                             const _Float16* __restrict__ clus_h,
                                             const _Float16* __restrict__ wdot_h,
                                             const float* __restrict__ b_dot,
                                             float* __restrict__ q_ws,
                                             float* __restrict__ key_ws) {
    const int tile = blockIdx.x;
    const int m    = blockIdx.y;
    const int lane = threadIdx.x;
    const int l    = l_of_m(m);
    int row0 = tile * 16;
    const _Float16* src;
    float* dst;
    int base;
    if (row0 < NN) { src = node_h; dst = q_ws;   base = row0; }
    else           { src = clus_h; dst = key_ws; base = row0 - NN; }
    const _Float16* Wl   = wdot_h + (size_t)l * AWW * CCd;
    const _Float16* arow = src + (size_t)base * 9 * CCd + (size_t)m * CCd;

    for (int nt = 0; nt < 2; ++nt) {
        v8f c = {};
        for (int kt = 0; kt < 8; ++kt) {
            v16h a = load_a_h_full(arow, 9 * CCd, 0, kt * 32, lane);
            v16h b = load_b_h_full(Wl, CCd, nt * 16, kt * 32, lane);
            c = wmma16(a, b, c);
        }
        const int radd = (lane & 16) ? 8 : 0;
        const int d    = nt * 16 + (lane & 15);
        const float bias = (l == 0) ? b_dot[d] : 0.0f;
#pragma unroll
        for (int r = 0; r < 8; ++r)
            dst[(size_t)(base + r + radd) * 9 * AWW + m * AWW + d] = c[r] + bias;
    }
}

// ---- Kernel 2: per-node attention -------------------------------------------
struct MainParams {
    const float *edge_vec, *attn_w, *tgt_emb, *cluster_ir;
    const int *atomic_num, *sp_idx;
    const unsigned char *attn_mask;
    const float *q_ws, *key_ws;
    // f16 weights
    const _Float16 *rm0w1_h, *rm0w2_h, *rm0w3_h, *rvw1_h, *rvw2_h, *rvw3_h, *fc_h;
    // f32 biases / LN params
    const float *rm0_b1, *rm0_g1, *rm0_be1, *rm0_b2, *rm0_g2, *rm0_be2, *rm0_b3;
    const float *fc_b, *an_g, *an_b, *alpha_dot;
    const float *rv_b1, *rv_g1, *rv_be1, *rv_b2, *rv_g2, *rv_be2, *rv_b3;
    _Float16 *pre_out;
};

__device__ __forceinline__ void ln_silu_row16(const float* h, _Float16* out,
                                              const float* bb, const float* gg,
                                              const float* be) {
    float v[16], mu = 0.0f;
#pragma unroll
    for (int i = 0; i < 16; ++i) { v[i] = h[i] + bb[i]; mu += v[i]; }
    mu *= (1.0f / 16.0f);
    float var = 0.0f;
#pragma unroll
    for (int i = 0; i < 16; ++i) { float d = v[i] - mu; var += d * d; }
    var *= (1.0f / 16.0f);
    float rstd = rsqrtf(var + 1e-5f);
#pragma unroll
    for (int i = 0; i < 16; ++i) {
        float x = (v[i] - mu) * rstd * gg[i] + be[i];
        out[i] = (_Float16)(x / (1.0f + expf(-x)));   // SiLU
    }
}

__global__ __launch_bounds__(256) void k_main(MainParams P) {
    const int n    = blockIdx.x;
    const int t    = threadIdx.x;
    const int lane = t & 31;
    const int wid  = t >> 5;

    __shared__ __align__(32) _Float16 s_xe[KK * ECC];   // x_edge (A operand)
    __shared__ __align__(32) _Float16 s_m0h[KK * M0D];  // (m0*ew) f16 (A operand)
    __shared__ float s_big[KK * CCd];                   // af, then vw
    __shared__ float s_m0[KK * M0D];                    // m0 (f32)
    __shared__ float s_q[9 * AWW];
    __shared__ float s_sh[KK * 9];
    __shared__ float s_h1[KK * 16], s_h1v[KK * 16];     // layer1 pre-act
    __shared__ float s_h2[KK * 16], s_h2v[KK * 16];     // layer2 pre-act
    __shared__ __align__(32) _Float16 s_h1h[KK * 16], s_h1vh[KK * 16];
    __shared__ __align__(32) _Float16 s_h2h[KK * 16], s_h2vh[KK * 16];
    __shared__ float s_logit[KK * 8], s_alpha[KK * 8];
    __shared__ int   s_idx[KK];
    __shared__ float s_mask[KK];
    __shared__ float s_tgt[NEE];

    // ---- A1: indices, mask, spherical harmonics, q, tgt ----
    if (t < KK) {
        s_idx[t]  = P.sp_idx[n * KK + t];
        s_mask[t] = P.attn_mask[n * KK + t] ? 1.0f : 0.0f;
        const float* ev = P.edge_vec + (size_t)(n * KK + t) * 3;
        float x = ev[0], y = ev[1], z = ev[2];
        float inv = 1.0f / (sqrtf(x * x + y * y + z * z) + 1e-12f);
        x *= inv; y *= inv; z *= inv;
        float* sh = s_sh + t * 9;
        sh[0] = 0.28209479177387814f;
        sh[1] = 0.4886025119029199f * y;
        sh[2] = 0.4886025119029199f * z;
        sh[3] = 0.4886025119029199f * x;
        sh[4] = 1.0925484305920792f * x * y;
        sh[5] = 1.0925484305920792f * y * z;
        sh[6] = 0.31539156525252005f * (3.0f * z * z - 1.0f);
        sh[7] = 1.0925484305920792f * x * z;
        sh[8] = 0.5462742152960396f * (x * x - y * y);
    }
    for (int j = t; j < 9 * AWW; j += 256) s_q[j] = P.q_ws[(size_t)n * 9 * AWW + j];
    if (t < NEE) {
        int an = P.atomic_num[n];
        s_tgt[t] = P.tgt_emb[(size_t)an * NEE + t];
    }
    __syncthreads();

    // ---- A2: x_edge = [aw(masked) | tgt | cluster l=0 scalars], f16 ----
    for (int j = t; j < KK * ECC; j += 256) {
        int k = j / ECC, c = j % ECC;
        float v;
        if (c < AWW)
            v = (s_mask[k] > 0.5f) ? 0.0f : P.attn_w[((size_t)n * KK + k) * AWW + c];
        else if (c < AWW + NEE)
            v = s_tgt[c - AWW];
        else
            v = P.cluster_ir[(size_t)s_idx[k] * 9 * CCd + (c - AWW - NEE)];
        s_xe[j] = (_Float16)v;
    }
    // ---- B: m0 degree-dot features (q part cols 0..95, key part 96..191) ----
    {
        const int s0[3] = {0, 1, 4}, e0[3] = {1, 4, 9};
        for (int j = t; j < KK * M0D; j += 256) {
            int k = j / M0D, col = j % M0D;
            float acc = 0.0f;
            if (col < 96) {
                int l = col >> 5, cc = col & 31;
                for (int m = s0[l]; m < e0[l]; ++m)
                    acc += s_q[m * 32 + cc] * s_sh[k * 9 + m];
            } else {
                int col2 = col - 96;
                int l = col2 >> 5, cc = col2 & 31;
                const float* kr = P.key_ws + (size_t)s_idx[k] * 9 * AWW;
                for (int m = s0[l]; m < e0[l]; ++m)
                    acc += kr[m * 32 + cc] * s_sh[k * 9 + m];
            }
            s_m0[j] = acc;
        }
    }
    __syncthreads();

    // ---- C: radial layer1 (both MLPs in parallel, waves 0..3) ----
    if (wid < 4) {
        const int  mt   = wid & 1;
        const bool is_v = (wid >= 2);
        const _Float16* w1 = is_v ? P.rvw1_h : P.rm0w1_h;
        float* dst         = is_v ? s_h1v : s_h1;
        v8f c = {};
        for (int kt = 0; kt < 13; ++kt) {
            v16h a = load_a_h_full(s_xe, ECC, mt * 16, kt * 32, lane);
            v16h b = load_b_h_full(w1, ECC, 0, kt * 32, lane);
            c = wmma16(a, b, c);
        }
        const int radd = (lane & 16) ? 8 : 0;
        const int col  = lane & 15;
#pragma unroll
        for (int r = 0; r < 8; ++r) dst[(mt * 16 + r + radd) * 16 + col] = c[r];
    }
    __syncthreads();
    if (t < 64) {
        const int  k    = t & 31;
        const bool is_v = (t >= 32);
        ln_silu_row16((is_v ? s_h1v : s_h1) + k * 16, (is_v ? s_h1vh : s_h1h) + k * 16,
                      is_v ? P.rv_b1 : P.rm0_b1, is_v ? P.rv_g1 : P.rm0_g1,
                      is_v ? P.rv_be1 : P.rm0_be1);
    }
    __syncthreads();

    // ---- D: radial layer2 (K=16 padded to 32) ----
    if (wid < 4) {
        const int  mt   = wid & 1;
        const bool is_v = (wid >= 2);
        const _Float16* w2   = is_v ? P.rvw2_h : P.rm0w2_h;
        const _Float16* hsrc = is_v ? s_h1vh : s_h1h;
        float* dst           = is_v ? s_h2v : s_h2;
        v16h a = load_a_h_pad(hsrc, 16, mt * 16, 16, lane);
        v16h b = load_b_h_pad(w2, 16, 0, 16, lane);
        v8f c = {};
        c = wmma16(a, b, c);
        const int radd = (lane & 16) ? 8 : 0;
        const int col  = lane & 15;
#pragma unroll
        for (int r = 0; r < 8; ++r) dst[(mt * 16 + r + radd) * 16 + col] = c[r];
    }
    __syncthreads();
    if (t < 64) {
        const int  k    = t & 31;
        const bool is_v = (t >= 32);
        ln_silu_row16((is_v ? s_h2v : s_h2) + k * 16, (is_v ? s_h2vh : s_h2h) + k * 16,
                      is_v ? P.rv_b2 : P.rm0_b2, is_v ? P.rv_g2 : P.rm0_g2,
                      is_v ? P.rv_be2 : P.rm0_be2);
    }
    __syncthreads();

    // ---- E: ew = h2 @ rm0_w3^T + b3 ; s_m0h = f16(m0 * ew) ----
    for (int job = wid * 3; job < wid * 3 + 3; ++job) {   // 24 jobs = 2 Mt x 12 Nt
        const int mt = job & 1, nt = job >> 1;
        v16h a = load_a_h_pad(s_h2h, 16, mt * 16, 16, lane);
        v16h b = load_b_h_pad(P.rm0w3_h, 16, nt * 16, 16, lane);
        v8f c = {};
        c = wmma16(a, b, c);
        const int radd = (lane & 16) ? 8 : 0;
        const int col  = nt * 16 + (lane & 15);
        const float b3 = P.rm0_b3[col];
#pragma unroll
        for (int r = 0; r < 8; ++r) {
            int row = mt * 16 + r + radd;
            s_m0h[row * M0D + col] = (_Float16)(s_m0[row * M0D + col] * (c[r] + b3));
        }
    }
    __syncthreads();

    // ---- F: af = (m0*ew) @ fc_m0_w^T + fc_b  -> s_big [32 x 256] ----
    for (int job = wid * 4; job < wid * 4 + 4; ++job) {   // 32 jobs = 2 Mt x 16 Nt
        const int mt = job & 1, nt = job >> 1;
        v8f c = {};
        for (int kt = 0; kt < 6; ++kt) {
            v16h a = load_a_h_full(s_m0h, M0D, mt * 16, kt * 32, lane);
            v16h b = load_b_h_full(P.fc_h, M0D, nt * 16, kt * 32, lane);
            c = wmma16(a, b, c);
        }
        const int radd = (lane & 16) ? 8 : 0;
        const int col  = nt * 16 + (lane & 15);
        const float bb = P.fc_b[col];
#pragma unroll
        for (int r = 0; r < 8; ++r)
            s_big[(mt * 16 + r + radd) * CCd + col] = c[r] + bb;
    }
    __syncthreads();

    // ---- G: LN over S + smooth-leaky + logits, one thread per (k,h) ----
    {
        const int k = t >> 3, h = t & 7;
        float* row = s_big + k * CCd + h * SSs;
        float mu = 0.0f;
        for (int s = 0; s < SSs; ++s) mu += row[s];
        mu *= (1.0f / SSs);
        float var = 0.0f;
        for (int s = 0; s < SSs; ++s) { float d = row[s] - mu; var += d * d; }
        var *= (1.0f / SSs);
        float rstd = rsqrtf(var + 1e-5f);
        float acc = 0.0f;
        for (int s = 0; s < SSs; ++s) {
            float x  = (row[s] - mu) * rstd * P.an_g[s] + P.an_b[s];
            float sg = 1.0f / (1.0f + expf(-x));
            float y  = 0.6f * x + 0.4f * x * (2.0f * sg - 1.0f);   // smooth_leaky a=0.2
            acc += y * P.alpha_dot[h * SSs + s];
        }
        s_logit[k * 8 + h] = (s_mask[k] > 0.5f) ? -1e9f : acc;
    }
    __syncthreads();
    if (t < 8) {   // softmax over K per head
        float mx = -3.0e38f;
        for (int k = 0; k < KK; ++k) mx = fmaxf(mx, s_logit[k * 8 + t]);
        float sum = 0.0f;
        for (int k = 0; k < KK; ++k) sum += expf(s_logit[k * 8 + t] - mx);
        float inv = 1.0f / sum;
        for (int k = 0; k < KK; ++k)
            s_alpha[k * 8 + t] = expf(s_logit[k * 8 + t] - mx) * inv;
    }
    __syncthreads();

    // ---- H: vw = h2v @ rv_w3^T + b3 -> s_big (af no longer needed) ----
    for (int job = wid * 4; job < wid * 4 + 4; ++job) {
        const int mt = job & 1, nt = job >> 1;
        v16h a = load_a_h_pad(s_h2vh, 16, mt * 16, 16, lane);
        v16h b = load_b_h_pad(P.rvw3_h, 16, nt * 16, 16, lane);
        v8f c = {};
        c = wmma16(a, b, c);
        const int radd = (lane & 16) ? 8 : 0;
        const int col  = nt * 16 + (lane & 15);
        const float b3 = P.rv_b3[col];
#pragma unroll
        for (int r = 0; r < 8; ++r)
            s_big[(mt * 16 + r + radd) * CCd + col] = c[r] + b3;
    }
    __syncthreads();

    // ---- I: out[m,c] = sum_k alpha[k,h(c)] * cluster[idx[k],m,c] * vw[k,c] ----
    {
        const int c = t;
        const int h = c >> 5;
        _Float16* po = P.pre_out + (size_t)n * 9 * CCd;
        for (int m = 0; m < 9; ++m) {
            float acc = 0.0f;
            for (int k = 0; k < KK; ++k) {
                acc += s_alpha[k * 8 + h] *
                       P.cluster_ir[(size_t)s_idx[k] * 9 * CCd + m * CCd + c] *
                       s_big[k * CCd + c];
            }
            po[m * CCd + c] = (_Float16)acc;
        }
    }
}

// ---- Kernel 3: final so3_linear with wz/bz ----------------------------------
// grid: (128, 9)  block: 128 (4 waves, 4 N-tiles each). A tiles cached in regs.
__global__ __launch_bounds__(128) void k_final(const _Float16* __restrict__ pre_h,
                                               const _Float16* __restrict__ wz_h,
                                               const float* __restrict__ bz,
                                               float* __restrict__ out) {
    const int tile = blockIdx.x;
    const int m    = blockIdx.y;
    const int t    = threadIdx.x;
    const int lane = t & 31, wid = t >> 5;
    const int l    = l_of_m(m);
    __shared__ __align__(32) _Float16 s_pre[16 * CCd];
    const int n0 = tile * 16;
    for (int j = t; j < 16 * CCd; j += 128) {
        int r = j >> 8, c = j & 255;
        s_pre[j] = pre_h[((size_t)(n0 + r) * 9 + m) * CCd + c];
    }
    __syncthreads();
    v16h aReg[8];
#pragma unroll
    for (int kt = 0; kt < 8; ++kt) aReg[kt] = load_a_h_full(s_pre, CCd, 0, kt * 32, lane);
    const _Float16* Wl = wz_h + (size_t)l * CCd * CCd;
    for (int i = 0; i < 4; ++i) {
        const int nt = wid * 4 + i;
        v8f c = {};
#pragma unroll
        for (int kt = 0; kt < 8; ++kt) {
            v16h b = load_b_h_full(Wl, CCd, nt * 16, kt * 32, lane);
            c = wmma16(aReg[kt], b, c);
        }
        const int radd = (lane & 16) ? 8 : 0;
        const int d    = nt * 16 + (lane & 15);
        const float bias = (l == 0) ? bz[d] : 0.0f;
#pragma unroll
        for (int r = 0; r < 8; ++r)
            out[((size_t)(n0 + r + radd) * 9 + m) * CCd + d] = c[r] + bias;
    }
}

// ---- launcher ---------------------------------------------------------------
extern "C" void kernel_launch(void* const* d_in, const int* in_sizes, int n_in,
                              void* d_out, int out_size, void* d_ws, size_t ws_size,
                              hipStream_t stream) {
    (void)in_sizes; (void)n_in; (void)out_size; (void)ws_size;
    const float* node_ir    = (const float*)d_in[1];
    const float* edge_vec   = (const float*)d_in[3];
    const float* attn_w     = (const float*)d_in[4];
    const int*   atomic_num = (const int*)d_in[5];
    const unsigned char* attn_mask = (const unsigned char*)d_in[6];
    const int*   sp_idx     = (const int*)d_in[7];
    const float* cluster_ir = (const float*)d_in[8];
    const float* tgt_emb    = (const float*)d_in[9];
    const float* W_dot      = (const float*)d_in[10];
    const float* b_dot      = (const float*)d_in[11];
    const float* wz         = (const float*)d_in[37];
    const float* bz         = (const float*)d_in[38];

    // ---- workspace carve-out (256B aligned) ----
    char* wsb = (char*)d_ws;
    size_t off = 0;
    auto carve = [&](size_t bytes) -> void* {
        void* p = wsb + off;
        off = (off + bytes + 255) & ~(size_t)255;
        return p;
    };
    float*     q_ws    = (float*)carve((size_t)NN * 9 * AWW * 4);
    float*     key_ws  = (float*)carve((size_t)MMc * 9 * AWW * 4);
    _Float16*  pre_h   = (_Float16*)carve((size_t)NN * 9 * CCd * 2);
    _Float16*  wdot_h  = (_Float16*)carve((size_t)3 * AWW * CCd * 2);
    _Float16*  rm0w1_h = (_Float16*)carve((size_t)HIDD * ECC * 2);
    _Float16*  rm0w2_h = (_Float16*)carve((size_t)HIDD * HIDD * 2);
    _Float16*  rm0w3_h = (_Float16*)carve((size_t)M0D * HIDD * 2);
    _Float16*  rvw1_h  = (_Float16*)carve((size_t)HIDD * ECC * 2);
    _Float16*  rvw2_h  = (_Float16*)carve((size_t)HIDD * HIDD * 2);
    _Float16*  rvw3_h  = (_Float16*)carve((size_t)CCd * HIDD * 2);
    _Float16*  fc_h    = (_Float16*)carve((size_t)CCd * M0D * 2);
    _Float16*  wz_h    = (_Float16*)carve((size_t)3 * CCd * CCd * 2);
    _Float16*  node_h  = (_Float16*)carve((size_t)NN * 9 * CCd * 2);
    _Float16*  clus_h  = (_Float16*)carve((size_t)MMc * 9 * CCd * 2);

    auto cvt = [&](const float* s, _Float16* d, size_t n) {
        k_cvt<<<dim3((unsigned)((n + 255) / 256)), 256, 0, stream>>>(s, d, (int)n);
    };
    cvt(W_dot, wdot_h, (size_t)3 * AWW * CCd);
    cvt((const float*)d_in[12], rm0w1_h, (size_t)HIDD * ECC);
    cvt((const float*)d_in[16], rm0w2_h, (size_t)HIDD * HIDD);
    cvt((const float*)d_in[20], rm0w3_h, (size_t)M0D * HIDD);
    cvt((const float*)d_in[27], rvw1_h, (size_t)HIDD * ECC);
    cvt((const float*)d_in[31], rvw2_h, (size_t)HIDD * HIDD);
    cvt((const float*)d_in[35], rvw3_h, (size_t)CCd * HIDD);
    cvt((const float*)d_in[22], fc_h, (size_t)CCd * M0D);
    cvt(wz, wz_h, (size_t)3 * CCd * CCd);
    cvt(node_ir, node_h, (size_t)NN * 9 * CCd);
    cvt(cluster_ir, clus_h, (size_t)MMc * 9 * CCd);

    dim3 g1((NN + MMc) / 16, 9);
    k_proj<<<g1, 32, 0, stream>>>(node_h, clus_h, wdot_h, b_dot, q_ws, key_ws);

    MainParams P;
    P.edge_vec = edge_vec; P.attn_w = attn_w; P.tgt_emb = tgt_emb; P.cluster_ir = cluster_ir;
    P.atomic_num = atomic_num; P.sp_idx = sp_idx; P.attn_mask = attn_mask;
    P.q_ws = q_ws; P.key_ws = key_ws;
    P.rm0w1_h = rm0w1_h; P.rm0w2_h = rm0w2_h; P.rm0w3_h = rm0w3_h;
    P.rvw1_h = rvw1_h; P.rvw2_h = rvw2_h; P.rvw3_h = rvw3_h; P.fc_h = fc_h;
    P.rm0_b1 = (const float*)d_in[13]; P.rm0_g1 = (const float*)d_in[14];
    P.rm0_be1 = (const float*)d_in[15];
    P.rm0_b2 = (const float*)d_in[17]; P.rm0_g2 = (const float*)d_in[18];
    P.rm0_be2 = (const float*)d_in[19];
    P.rm0_b3 = (const float*)d_in[21];
    P.fc_b = (const float*)d_in[23];
    P.an_g = (const float*)d_in[24]; P.an_b = (const float*)d_in[25];
    P.alpha_dot = (const float*)d_in[26];
    P.rv_b1 = (const float*)d_in[28]; P.rv_g1 = (const float*)d_in[29];
    P.rv_be1 = (const float*)d_in[30];
    P.rv_b2 = (const float*)d_in[32]; P.rv_g2 = (const float*)d_in[33];
    P.rv_be2 = (const float*)d_in[34];
    P.rv_b3 = (const float*)d_in[36];
    P.pre_out = pre_h;
    k_main<<<NN, 256, 0, stream>>>(P);

    dim3 g3(NN / 16, 9);
    k_final<<<g3, 128, 0, stream>>>(pre_h, wz_h, bz, (float*)d_out);
}
